// MultiHeadAttention_32822140076457
// MI455X (gfx1250) — matrix operand order repaired and checked by reading, not compiled, not run on hardware
//
#include <hip/hip_runtime.h>

// ---------------------------------------------------------------------------
// MultiHeadAttention forward for MI455X (gfx1250, wave32, WMMA bf16,
// async global->LDS double buffering in the attention inner loop)
// ---------------------------------------------------------------------------

typedef __attribute__((ext_vector_type(16))) __bf16 v16bf;
typedef __attribute__((ext_vector_type(8)))  float  v8f;
typedef __attribute__((ext_vector_type(8)))  unsigned short ushort8;
typedef __attribute__((ext_vector_type(4)))  unsigned short ushort4v;
typedef __attribute__((ext_vector_type(4)))  float  float4v;

union Frag16 {
  v16bf   bf;
  ushort8 h[2];
};

#define B_  2
#define S_  2048
#define D_  1024
#define H_  16
#define DH_ 64

__device__ __forceinline__ unsigned short f32_to_bf16_rne(float f) {
  union { float f; unsigned u; } x; x.f = f;
  unsigned u = x.u;
  unsigned r = u + 0x7FFFu + ((u >> 16) & 1u);
  return (unsigned short)(r >> 16);
}

__device__ __forceinline__ v8f wmma_bf16(const Frag16& a, const Frag16& b, v8f c) {
  return __builtin_amdgcn_wmma_f32_16x16x32_bf16(false, a.bf, false, b.bf,
                                                 (short)0, c, false, false);
}

// Async copy of one 16-byte chunk global -> LDS (ASYNCcnt-tracked TDM-lite path).
__device__ __forceinline__ void async_load_b128(unsigned lds_off, const void* gptr) {
  unsigned long long ga = (unsigned long long)(uintptr_t)gptr;
  asm volatile("global_load_async_to_lds_b128 %0, %1, off"
               :: "v"(lds_off), "v"(ga) : "memory");
}
__device__ __forceinline__ void wait_async0() {
  asm volatile("s_wait_asynccnt 0x0" ::: "memory");
}

// ---------------------------------------------------------------------------
// f32 -> bf16 bulk conversion (vectorized: float4 -> 4 x bf16)
// ---------------------------------------------------------------------------
__global__ void cvt_f32_bf16_kernel(const float4v* __restrict__ in,
                                    ushort4v* __restrict__ out, int n4) {
  int i = blockIdx.x * blockDim.x + threadIdx.x;
  if (i < n4) {
    float4v f = in[i];
    ushort4v o;
    o.x = f32_to_bf16_rne(f.x);
    o.y = f32_to_bf16_rne(f.y);
    o.z = f32_to_bf16_rne(f.z);
    o.w = f32_to_bf16_rne(f.w);
    out[i] = o;
  }
}

// ---------------------------------------------------------------------------
// GEMM: C(M x N) = A(M x K) @ W^T + bias, A/W bf16 row-major, f32 accumulate.
// One wave computes a 32x32 macro-tile: 2 A-frags x 2 B-frags -> 4 WMMAs per
// 32-wide k-step (2 b128 loads per WMMA).
// mode 0: f32 out, row-major (M x N)                        (final projection)
// mode 1: bf16 out, head layout    [(b*H+h)*S + s]*DH + di  (Q, K)
// mode 2: bf16 out, head-T layout  [(b*H+h)*DH + di]*S + s  (V transposed)
// ---------------------------------------------------------------------------
__device__ __forceinline__ void store_tile(v8f acc, int mBase, int n, int rb,
                                           float bvv, int N, int mode,
                                           float* __restrict__ outF,
                                           unsigned short* __restrict__ outH) {
#pragma unroll
  for (int r = 0; r < 8; ++r) {
    const int m = mBase + rb + r;
    const float v = acc[r] + bvv;
    if (mode == 0) {
      outF[(size_t)m * N + n] = v;
    } else {
      const int h  = n / DH_;
      const int di = n % DH_;
      const int bb = m / S_;
      const int s  = m % S_;
      const unsigned short hv = f32_to_bf16_rne(v);
      if (mode == 1) outH[((size_t)(bb * H_ + h) * S_ + s) * DH_ + di] = hv;
      else           outH[((size_t)(bb * H_ + h) * DH_ + di) * S_ + s] = hv;
    }
  }
}

__global__ void gemm_xwT_kernel(const unsigned short* __restrict__ A,
                                const unsigned short* __restrict__ W,
                                const float* __restrict__ bias,
                                float* __restrict__ outF,
                                unsigned short* __restrict__ outH,
                                int M, int K, int N, int mode) {
  const int lane = threadIdx.x & 31;
  const int col  = lane & 15;
  const int half = lane >> 4;
  const int n0   = blockIdx.x * 32;
  const int m0   = blockIdx.y * 32;

  const int akb = half * 8;    // A-frag K base within 32-step
  const int bkb = half * 8;    // B-frag K base within 32-step

  const unsigned short* aRow0 = A + (size_t)(m0 + col) * K;
  const unsigned short* aRow1 = A + (size_t)(m0 + 16 + col) * K;
  const unsigned short* wRow0 = W + (size_t)(n0 + col) * K;
  const unsigned short* wRow1 = W + (size_t)(n0 + 16 + col) * K;

  v8f c00 = {}, c01 = {}, c10 = {}, c11 = {};
  for (int k0 = 0; k0 < K; k0 += 32) {
    Frag16 a0, a1, b0, b1;
    const unsigned short* ap0 = aRow0 + k0 + akb;
    const unsigned short* ap1 = aRow1 + k0 + akb;
    const unsigned short* bp0 = wRow0 + k0 + bkb;
    const unsigned short* bp1 = wRow1 + k0 + bkb;
    a0.h[0] = *(const ushort8*)(ap0); a0.h[1] = *(const ushort8*)(ap0 + 16);
    a1.h[0] = *(const ushort8*)(ap1); a1.h[1] = *(const ushort8*)(ap1 + 16);
    b0.h[0] = *(const ushort8*)(bp0); b0.h[1] = *(const ushort8*)(bp0 + 16);
    b1.h[0] = *(const ushort8*)(bp1); b1.h[1] = *(const ushort8*)(bp1 + 16);
    __builtin_prefetch(ap0 + 64, 0, 0);
    __builtin_prefetch(ap1 + 64, 0, 0);
    __builtin_prefetch(bp0 + 64, 0, 0);
    __builtin_prefetch(bp1 + 64, 0, 0);
    c00 = wmma_bf16(a0, b0, c00);
    c01 = wmma_bf16(a0, b1, c01);
    c10 = wmma_bf16(a1, b0, c10);
    c11 = wmma_bf16(a1, b1, c11);
  }

  const float bv0 = bias[n0 + col];
  const float bv1 = bias[n0 + 16 + col];
  const int   rb  = half * 8;
  store_tile(c00, m0,      n0 + col,      rb, bv0, N, mode, outF, outH);
  store_tile(c01, m0,      n0 + 16 + col, rb, bv1, N, mode, outF, outH);
  store_tile(c10, m0 + 16, n0 + col,      rb, bv0, N, mode, outF, outH);
  store_tile(c11, m0 + 16, n0 + 16 + col, rb, bv1, N, mode, outF, outH);
}

// ---------------------------------------------------------------------------
// Flash-style attention, one wave per (b,h, 16 query rows).
// K/V 32-key blocks are double-buffered in LDS via global_load_async_to_lds,
// overlapping next-block fetch (ASYNCcnt) with current-block WMMAs.
// ---------------------------------------------------------------------------
__global__ void attn_kernel(const unsigned short* __restrict__ Q,   // (B*H,S,DH)
                            const unsigned short* __restrict__ Km,  // (B*H,S,DH)
                            const unsigned short* __restrict__ Vt,  // (B*H,DH,S)
                            const int* __restrict__ mask,           // (B,S,S)
                            unsigned short* __restrict__ O,         // (B,S,D) bf16
                            float scale) {
  __shared__ __align__(16) unsigned short Klds[2][32 * DH_];  // 2 x 4KB
  __shared__ __align__(16) unsigned short Vlds[2][DH_ * 32];  // 2 x 4KB
  __shared__ __align__(16) unsigned short Plds[16 * 32];      // 1KB

  const int lane = threadIdx.x & 31;
  const int col  = lane & 15;
  const int half = lane >> 4;
  const int rb   = half * 8;
  const int bh   = blockIdx.y;
  const int b    = bh / H_;
  const int h    = bh % H_;
  const int q0   = blockIdx.x * 16;

  const unsigned short* Qb = Q  + (size_t)bh * S_ * DH_;
  const unsigned short* Kb = Km + (size_t)bh * S_ * DH_;
  const unsigned short* Vb = Vt + (size_t)bh * DH_ * S_;

  // Issue async fetch of one 32-key K/V block into LDS buffer `buf`.
  auto fetch_block = [&](int kb, int buf) {
    // K block: 32 rows x 128B = 256 x 16B chunks, 8 per lane.
#pragma unroll
    for (int c = 0; c < 8; ++c) {
      const int idx = c * 32 + lane;
      const int row = idx >> 3;              // 0..31
      const int bo  = (idx & 7) * 16;        // byte offset in row
      const char* g = (const char*)(Kb + (size_t)(kb + row) * DH_) + bo;
      const unsigned l = (unsigned)(uintptr_t)((const char*)&Klds[buf][row * DH_] + bo);
      async_load_b128(l, g);
    }
    // V block (transposed layout): 64 rows x 64B = 256 x 16B chunks, 8 per lane.
#pragma unroll
    for (int c = 0; c < 8; ++c) {
      const int idx = c * 32 + lane;
      const int row = idx >> 2;              // 0..63
      const int bo  = (idx & 3) * 16;
      const char* g = (const char*)(Vb + (size_t)row * S_ + kb) + bo;
      const unsigned l = (unsigned)(uintptr_t)((const char*)&Vlds[buf][row * 32] + bo);
      async_load_b128(l, g);
    }
  };

  // Q A-fragments for full head dim (two K-steps of 32), resident all loop.
  Frag16 qa0, qa1;
  {
    const unsigned short* qp = Qb + (size_t)(q0 + col) * DH_ + half * 8;
    qa0.h[0] = *(const ushort8*)(qp);
    qa0.h[1] = *(const ushort8*)(qp + 16);
    qa1.h[0] = *(const ushort8*)(qp + 32);
    qa1.h[1] = *(const ushort8*)(qp + 48);
  }

  float m_row[8], l_row[8];
  v8f o_acc[4];
#pragma unroll
  for (int r = 0; r < 8; ++r) { m_row[r] = -3.0e38f; l_row[r] = 0.0f; }
#pragma unroll
  for (int t = 0; t < 4; ++t) o_acc[t] = (v8f){};

  fetch_block(0, 0);
  wait_async0();
  __syncthreads();

  int buf = 0;
  for (int kb = 0; kb < S_; kb += 32) {
    // Prefetch next block into the other buffer while we compute.
    if (kb + 32 < S_) fetch_block(kb + 32, buf ^ 1);

    // ---- scores: two 16x16 tiles over keys kb..kb+31 (K frags from LDS) ----
    v8f sc[2];
#pragma unroll
    for (int nt = 0; nt < 2; ++nt) {
      const unsigned short* kp = &Klds[buf][(nt * 16 + col) * DH_ + half * 8];
      Frag16 b0, b1;
      b0.h[0] = *(const ushort8*)(kp);        // d = 0..31 slice
      b0.h[1] = *(const ushort8*)(kp + 16);
      b1.h[0] = *(const ushort8*)(kp + 32);   // d = 32..63 slice
      b1.h[1] = *(const ushort8*)(kp + 48);
      v8f c = {};
      c = wmma_bf16(qa0, b0, c);
      c = wmma_bf16(qa1, b1, c);
      sc[nt] = c;
    }
    // ---- scale + mask ----
#pragma unroll
    for (int nt = 0; nt < 2; ++nt) {
#pragma unroll
      for (int r = 0; r < 8; ++r) {
        const int q  = q0 + rb + r;
        const int kk = kb + nt * 16 + col;
        const int mv = mask[((size_t)b * S_ + q) * S_ + kk];
        sc[nt][r] = (mv == 0) ? -1.0e9f : sc[nt][r] * scale;
      }
    }
    // ---- online softmax: row max (reductions stay within 16-lane halves) ----
    float corr[8];
#pragma unroll
    for (int r = 0; r < 8; ++r) {
      float v = fmaxf(sc[0][r], sc[1][r]);
      v = fmaxf(v, __shfl_xor(v, 1, 32));
      v = fmaxf(v, __shfl_xor(v, 2, 32));
      v = fmaxf(v, __shfl_xor(v, 4, 32));
      v = fmaxf(v, __shfl_xor(v, 8, 32));
      const float nm = fmaxf(m_row[r], v);
      corr[r] = __expf(m_row[r] - nm);
      m_row[r] = nm;
    }
    // ---- p = exp(s-m), row sums, stage P in LDS as bf16 ----
#pragma unroll
    for (int r = 0; r < 8; ++r) {
      const float p0 = __expf(sc[0][r] - m_row[r]);
      const float p1 = __expf(sc[1][r] - m_row[r]);
      Plds[(rb + r) * 32 + col]      = f32_to_bf16_rne(p0);
      Plds[(rb + r) * 32 + 16 + col] = f32_to_bf16_rne(p1);
      float rs = p0 + p1;
      rs += __shfl_xor(rs, 1, 32);
      rs += __shfl_xor(rs, 2, 32);
      rs += __shfl_xor(rs, 4, 32);
      rs += __shfl_xor(rs, 8, 32);
      l_row[r] = l_row[r] * corr[r] + rs;
    }
    __syncthreads();
    // ---- reload P in A-fragment layout (lane = row, K runs of 8) ----
    Frag16 pa;
    {
      const unsigned short* pp = &Plds[col * 32 + half * 8];
      pa.h[0] = *(const ushort8*)(pp);
      pa.h[1] = *(const ushort8*)(pp + 16);
    }
    // ---- O = O*corr + P @ V (4 d-tiles of 16, V frags from LDS) ----
#pragma unroll
    for (int t = 0; t < 4; ++t) {
      v8f c = o_acc[t];
#pragma unroll
      for (int r = 0; r < 8; ++r) c[r] *= corr[r];
      const unsigned short* vp = &Vlds[buf][(t * 16 + col) * 32 + half * 8];
      Frag16 vbf;
      vbf.h[0] = *(const ushort8*)(vp);
      vbf.h[1] = *(const ushort8*)(vp + 16);
      c = wmma_bf16(pa, vbf, c);
      o_acc[t] = c;
    }
    // Next block must be resident before we flip buffers.
    wait_async0();
    __syncthreads();
    buf ^= 1;
  }

  // ---- normalize and emit bf16 attention output in (B,S,D) layout ----
#pragma unroll
  for (int t = 0; t < 4; ++t) {
#pragma unroll
    for (int r = 0; r < 8; ++r) {
      const float v = o_acc[t][r] / l_row[r];
      const int s = q0 + rb + r;
      const int d = h * DH_ + t * 16 + col;
      O[((size_t)b * S_ + s) * D_ + d] = f32_to_bf16_rne(v);
    }
  }
}

// ---------------------------------------------------------------------------
// Host-side orchestration
// ---------------------------------------------------------------------------
extern "C" void kernel_launch(void* const* d_in, const int* in_sizes, int n_in,
                              void* d_out, int out_size, void* d_ws, size_t ws_size,
                              hipStream_t stream) {
  const float* xq_f  = (const float*)d_in[0];
  const float* xkv_f = (const float*)d_in[1];
  const int*   mask  = (const int*)d_in[2];
  const float* Wq_f  = (const float*)d_in[3];
  const float* bq    = (const float*)d_in[4];
  const float* Wk_f  = (const float*)d_in[5];
  const float* bk    = (const float*)d_in[6];
  const float* Wv_f  = (const float*)d_in[7];
  const float* bv    = (const float*)d_in[8];
  const float* Wo_f  = (const float*)d_in[9];
  const float* bo    = (const float*)d_in[10];
  float* out         = (float*)d_out;

  const size_t nX = (size_t)B_ * S_ * D_;   // 4,194,304
  const size_t nW = (size_t)D_ * D_;        // 1,048,576

  char* p = (char*)d_ws;
  unsigned short* xq_h  = (unsigned short*)p; p += nX * 2;
  unsigned short* xkv_h = (unsigned short*)p; p += nX * 2;
  unsigned short* Wq_h  = (unsigned short*)p; p += nW * 2;
  unsigned short* Wk_h  = (unsigned short*)p; p += nW * 2;
  unsigned short* Wv_h  = (unsigned short*)p; p += nW * 2;
  unsigned short* Wo_h  = (unsigned short*)p; p += nW * 2;
  unsigned short* Qh    = (unsigned short*)p; p += nX * 2;  // (B*H,S,DH)
  unsigned short* Kh    = (unsigned short*)p; p += nX * 2;  // (B*H,S,DH)
  unsigned short* Vth   = (unsigned short*)p; p += nX * 2;  // (B*H,DH,S)
  unsigned short* Ah    = (unsigned short*)p; p += nX * 2;  // attn out (B,S,D)

  const int cvtBlk = 256;
  cvt_f32_bf16_kernel<<<(int)(nX / 4 / cvtBlk), cvtBlk, 0, stream>>>(
      (const float4v*)xq_f,  (ushort4v*)xq_h,  (int)(nX / 4));
  cvt_f32_bf16_kernel<<<(int)(nX / 4 / cvtBlk), cvtBlk, 0, stream>>>(
      (const float4v*)xkv_f, (ushort4v*)xkv_h, (int)(nX / 4));
  cvt_f32_bf16_kernel<<<(int)(nW / 4 / cvtBlk), cvtBlk, 0, stream>>>(
      (const float4v*)Wq_f, (ushort4v*)Wq_h, (int)(nW / 4));
  cvt_f32_bf16_kernel<<<(int)(nW / 4 / cvtBlk), cvtBlk, 0, stream>>>(
      (const float4v*)Wk_f, (ushort4v*)Wk_h, (int)(nW / 4));
  cvt_f32_bf16_kernel<<<(int)(nW / 4 / cvtBlk), cvtBlk, 0, stream>>>(
      (const float4v*)Wv_f, (ushort4v*)Wv_h, (int)(nW / 4));
  cvt_f32_bf16_kernel<<<(int)(nW / 4 / cvtBlk), cvtBlk, 0, stream>>>(
      (const float4v*)Wo_f, (ushort4v*)Wo_h, (int)(nW / 4));

  const int M = B_ * S_;                 // 4096
  dim3 gGemm(D_ / 32, M / 32);           // (32, 128) 32x32 macro-tiles
  dim3 bWave(32);
  gemm_xwT_kernel<<<gGemm, bWave, 0, stream>>>(xq_h,  Wq_h, bq, nullptr, Qh,  M, D_, D_, 1);
  gemm_xwT_kernel<<<gGemm, bWave, 0, stream>>>(xkv_h, Wk_h, bk, nullptr, Kh,  M, D_, D_, 1);
  gemm_xwT_kernel<<<gGemm, bWave, 0, stream>>>(xkv_h, Wv_h, bv, nullptr, Vth, M, D_, D_, 2);

  dim3 gAttn(S_ / 16, B_ * H_);          // (128, 32)
  attn_kernel<<<gAttn, bWave, 0, stream>>>(Qh, Kh, Vth, mask, Ah, 0.125f);

  gemm_xwT_kernel<<<gGemm, bWave, 0, stream>>>(Ah, Wo_h, bo, out, nullptr, M, D_, D_, 0);

  (void)in_sizes; (void)n_in; (void)out_size; (void)ws_size;
}
